// ProteinInterfacePrediction_1597727834179
// MI455X (gfx1250) — matrix operand order repaired and checked by reading, not compiled, not run on hardware
//
#include <hip/hip_runtime.h>

typedef __attribute__((ext_vector_type(16))) __bf16 v16bf;
typedef __attribute__((ext_vector_type(8)))  __bf16 v8bf;
typedef __attribute__((ext_vector_type(8)))  float  v8f;

union AFrag { v16bf v; v8bf h[2]; };

#define TILE_W 66          // 64 output pixels + 2 halo
#define TILE_ELEMS (3*TILE_W*32)   // 6336 bf16
#define WB_ELEMS (9*2*32*16)       // 9216 bf16, B-fragment layout

// ---------------------------------------------------------------------------
// CDNA5 async global->LDS copy (ASYNCcnt path). GV mode: per-lane 64-bit
// global address; VDST VGPR holds the LDS byte offset (low 32 bits of the
// generic LDS address).
// ---------------------------------------------------------------------------
__device__ __forceinline__ void async_load16(void* lds_dst, const void* gsrc)
{
    unsigned lds_off = (unsigned)(uintptr_t)lds_dst;
    asm volatile("global_load_async_to_lds_b128 %0, %1, off"
                 :: "v"(lds_off), "v"(gsrc)
                 : "memory");
}
__device__ __forceinline__ void wait_async0()
{
    asm volatile("s_wait_asynccnt 0" ::: "memory");
}

// ---------------------------------------------------------------------------
// Stage 1: GNN (ligand + receptor fused). 4096 blocks (B*(L+R)), 128 threads.
// out = node + mean_k tanh(WN@node + bN + WE@edge_k + bE)
// ---------------------------------------------------------------------------
__global__ __launch_bounds__(128) void gnn_kernel(
    const float* __restrict__ lnode, const float* __restrict__ ledge,
    const float* __restrict__ rnode, const float* __restrict__ redge,
    const float* __restrict__ WN, const float* __restrict__ bN,
    const float* __restrict__ WE, const float* __restrict__ bE,
    float* __restrict__ lig, float* __restrict__ rec)
{
    __shared__ float s_node[128];
    __shared__ float s_edge[16 * 64];

    int r = blockIdx.x;
    int b = r >> 10;
    int idx = r & 1023;
    const float* node; const float* edge; float* outp;
    if (idx < 512) {
        node = lnode + ((size_t)(b * 512 + idx)) * 128;
        edge = ledge + ((size_t)(b * 512 + idx)) * (16 * 64);
        outp = lig   + ((size_t)(b * 512 + idx)) * 128;
    } else {
        int j = idx - 512;
        node = rnode + ((size_t)(b * 512 + j)) * 128;
        edge = redge + ((size_t)(b * 512 + j)) * (16 * 64);
        outp = rec   + ((size_t)(b * 512 + j)) * 128;
    }
    int t = threadIdx.x;
    s_node[t] = node[t];
    for (int q = t; q < 16 * 64; q += 128) s_edge[q] = edge[q];
    __syncthreads();

    const int o = t;
    const float* wn = WN + (size_t)o * 128;
    float hn = bN[o];
    #pragma unroll 4
    for (int d = 0; d < 128; ++d) hn += wn[d] * s_node[d];

    const float* we = WE + (size_t)o * 64;
    const float heb = bE[o];
    float acc = 0.f;
    for (int k = 0; k < 16; ++k) {
        float he = heb;
        #pragma unroll 4
        for (int d = 0; d < 64; ++d) he += we[d] * s_edge[k * 64 + d];
        acc += tanhf(hn + he);
    }
    outp[o] = s_node[o] + acc * (1.0f / 16.0f);
}

// ---------------------------------------------------------------------------
// Stage 2: pair projection. pl'[b,i,o] = Wl[o]·lig + bp[o] (bp folded here),
// pr[b,j,o] = Wr[o]·rec. 1024 blocks x 128 threads (4 rows/block).
// ---------------------------------------------------------------------------
__global__ __launch_bounds__(128) void pair_kernel(
    const float* __restrict__ lig, const float* __restrict__ rec,
    const float* __restrict__ Wp, const float* __restrict__ bp,
    float* __restrict__ plp, float* __restrict__ prr)
{
    __shared__ float s[4 * 128];
    int t = threadIdx.x;
    int sub = t >> 5, o = t & 31;
    int r = blockIdx.x * 4 + sub;            // 0..4095
    bool isRec = (r >= 2048);
    int rr = isRec ? (r - 2048) : r;
    const float* src = (isRec ? rec : lig) + (size_t)rr * 128;
    for (int q = o; q < 128; q += 32) s[sub * 128 + q] = src[q];
    __syncthreads();

    const float* w = Wp + (size_t)o * 256 + (isRec ? 128 : 0);
    float acc = isRec ? 0.f : bp[o];
    #pragma unroll 4
    for (int d = 0; d < 128; ++d) acc += w[d] * s[sub * 128 + d];
    (isRec ? prr : plp)[(size_t)rr * 32 + o] = acc;
}

// ---------------------------------------------------------------------------
// Weight reorder: Wc(32,32,3,3) fp32 -> bf16 WMMA B-fragment layout
// WB[tap][half][lane][e] = Wc[half*16 + (lane&15)][(lane>>4)*16 + e][di][dj]
// ---------------------------------------------------------------------------
__global__ void prep_w(const float* __restrict__ Wc, __bf16* __restrict__ WB)
{
    int q = blockIdx.x * blockDim.x + threadIdx.x;
    if (q >= WB_ELEMS) return;
    int e    = q & 15;
    int lane = (q >> 4) & 31;
    int half = (q >> 9) & 1;
    int tap  = q >> 10;
    int oc = half * 16 + (lane & 15);
    int c  = (lane >> 4) * 16 + e;
    int di = tap / 3, dj = tap % 3;
    WB[q] = (__bf16)Wc[((oc * 32 + c) * 3 + di) * 3 + dj];
}

// ---------------------------------------------------------------------------
// WMMA conv core: one wave computes a 16-pixel x 32-outchannel tile,
// K = 9 taps x 32 channels, 18 v_wmma_f32_16x16x32_bf16 per call.
// ---------------------------------------------------------------------------
__device__ __forceinline__ void conv_core(
    const __bf16* __restrict__ s_tile, const __bf16* __restrict__ s_wb,
    int lane, int wave, v8f& c0, v8f& c1)
{
    const int m  = lane & 15;
    const int hh = lane >> 4;
    #pragma unroll
    for (int tap = 0; tap < 9; ++tap) {
        const int di = tap / 3;
        const int dj = tap % 3;
        const __bf16* ap = s_tile + ((di * TILE_W) + (wave * 16 + m + dj)) * 32;
        AFrag a;
        a.h[0] = *(const v8bf*)(ap + hh * 16);       // channels h*16 .. h*16+7
        a.h[1] = *(const v8bf*)(ap + hh * 16 + 8);   // channels h*16+8 .. h*16+15
        const v16bf b0 = *(const v16bf*)(s_wb + ((tap * 2 + 0) * 32 + lane) * 16);
        const v16bf b1 = *(const v16bf*)(s_wb + ((tap * 2 + 1) * 32 + lane) * 16);
        c0 = __builtin_amdgcn_wmma_f32_16x16x32_bf16(false, a.v, false, b0,
                                                     (short)0, c0, false, false);
        c1 = __builtin_amdgcn_wmma_f32_16x16x32_bf16(false, a.v, false, b1,
                                                     (short)0, c1, false, false);
    }
}

// ---------------------------------------------------------------------------
// Conv1 (fused P construction): grid (8, 512, 4), 128 threads (4 waves).
// Builds P tile in LDS from pl'/pr (zero padded), WMMA conv, ReLU,
// LDS-repack, vectorized bf16 NHWC store of x1.
// ---------------------------------------------------------------------------
__global__ __launch_bounds__(128) void conv1_kernel(
    const float* __restrict__ plp, const float* __restrict__ prr,
    __bf16* __restrict__ x1,
    const __bf16* __restrict__ WBg, const float* __restrict__ bc1)
{
    __shared__ alignas(16) __bf16 s_tile[TILE_ELEMS];
    __shared__ alignas(32) __bf16 s_wb[WB_ELEMS];
    __shared__ float s_pl[3 * 32];
    __shared__ float s_pr[TILE_W * 32];

    const int jt = blockIdx.x, i = blockIdx.y, b = blockIdx.z;
    const int t = threadIdx.x;

    // weights -> LDS via async DMA path
    for (int q = t; q < WB_ELEMS / 8; q += 128)
        async_load16(&s_wb[q * 8], WBg + q * 8);

    for (int q = t; q < 3 * 32; q += 128) {
        int r = q >> 5, c = q & 31, ii = i - 1 + r;
        s_pl[q] = (ii >= 0 && ii < 512) ? plp[((size_t)(b * 512 + ii)) * 32 + c] : 0.f;
    }
    for (int q = t; q < TILE_W * 32; q += 128) {
        int jj = q >> 5, c = q & 31, j = jt * 64 - 1 + jj;
        s_pr[q] = (j >= 0 && j < 512) ? prr[((size_t)(b * 512 + j)) * 32 + c] : 0.f;
    }
    wait_async0();
    __syncthreads();
    for (int q = t; q < TILE_ELEMS; q += 128) {
        int r = q / (TILE_W * 32);
        int rem = q - r * (TILE_W * 32);
        int jj = rem >> 5, c = rem & 31;
        int ii = i - 1 + r, j = jt * 64 - 1 + jj;
        float v = (ii >= 0 && ii < 512 && j >= 0 && j < 512)
                      ? (s_pl[r * 32 + c] + s_pr[jj * 32 + c]) : 0.f;
        s_tile[q] = (__bf16)v;
    }
    __syncthreads();

    const int lane = t & 31, wave = t >> 5;
    v8f c0 = {0.f, 0.f, 0.f, 0.f, 0.f, 0.f, 0.f, 0.f};
    v8f c1 = {0.f, 0.f, 0.f, 0.f, 0.f, 0.f, 0.f, 0.f};
    conv_core(s_tile, s_wb, lane, wave, c0, c1);

    // repack D-fragments through LDS, then vectorized global stores
    __syncthreads();                 // all waves done reading s_tile
    const int n = lane & 15, hh = lane >> 4;
    const float b0 = bc1[n], b1 = bc1[16 + n];
    #pragma unroll
    for (int v = 0; v < 8; ++v) {
        int M = v + 8 * hh;
        float y0 = fmaxf(c0[v] + b0, 0.f);
        float y1 = fmaxf(c1[v] + b1, 0.f);
        s_tile[(wave * 16 + M) * 32 + n]      = (__bf16)y0;
        s_tile[(wave * 16 + M) * 32 + 16 + n] = (__bf16)y1;
    }
    __syncthreads();
    const uint4* sp = (const uint4*)s_tile;                 // 64*32 bf16 = 4KB
    uint4* gdst = (uint4*)(x1 + (((size_t)(b * 512 + i)) * 512 + jt * 64) * 32);
    for (int q = t; q < 256; q += 128) gdst[q] = sp[q];
}

// ---------------------------------------------------------------------------
// Conv2 + conv3(1x1) + sigmoid fused: grid (8, 512, 4), 128 threads.
// x1 halo tile -> LDS (async DMA, zero-filled OOB), WMMA conv, ReLU,
// per-pixel 32-ch dot via shfl_xor reduction, sigmoid, float4 stores.
// ---------------------------------------------------------------------------
__global__ __launch_bounds__(128) void conv2_kernel(
    const __bf16* __restrict__ x1, float* __restrict__ out,
    const __bf16* __restrict__ WBg, const float* __restrict__ bc2,
    const float* __restrict__ Wc3, const float* __restrict__ bc3)
{
    __shared__ alignas(16) __bf16 s_tile[TILE_ELEMS];
    __shared__ alignas(32) __bf16 s_wb[WB_ELEMS];

    const int jt = blockIdx.x, i = blockIdx.y, b = blockIdx.z;
    const int t = threadIdx.x;

    for (int q = t; q < WB_ELEMS / 8; q += 128)
        async_load16(&s_wb[q * 8], WBg + q * 8);

    uint4* tdst = (uint4*)s_tile;
    const int CHUNKS_PER_ROW = TILE_W * 32 / 8;        // 264
    for (int q = t; q < 3 * CHUNKS_PER_ROW; q += 128) {
        int r = q / CHUNKS_PER_ROW;
        int rem = q - r * CHUNKS_PER_ROW;
        int jj = rem >> 2;
        int cp = rem & 3;
        int ii = i - 1 + r, j = jt * 64 - 1 + jj;
        if (ii >= 0 && ii < 512 && j >= 0 && j < 512) {
            async_load16(&tdst[q],
                         x1 + ((((size_t)(b * 512 + ii)) * 512 + j) * 32 + cp * 8));
        } else {
            uint4 z = {0u, 0u, 0u, 0u};
            tdst[q] = z;
        }
    }
    wait_async0();
    __syncthreads();

    const int lane = t & 31, wave = t >> 5;
    v8f c0 = {0.f, 0.f, 0.f, 0.f, 0.f, 0.f, 0.f, 0.f};
    v8f c1 = {0.f, 0.f, 0.f, 0.f, 0.f, 0.f, 0.f, 0.f};
    conv_core(s_tile, s_wb, lane, wave, c0, c1);

    const int n = lane & 15, hh = lane >> 4;
    const float b0 = bc2[n], b1 = bc2[16 + n];
    const float w30 = Wc3[n], w31 = Wc3[16 + n];
    float s[8];
    #pragma unroll
    for (int v = 0; v < 8; ++v) {
        float y0 = fmaxf(c0[v] + b0, 0.f);
        float y1 = fmaxf(c1[v] + b1, 0.f);
        s[v] = w30 * y0 + w31 * y1;
    }
    // reduce over the 16 lanes of each half (xor masks 1,2,4,8 stay in-half)
    #pragma unroll
    for (int v = 0; v < 8; ++v) {
        float x = s[v];
        x += __shfl_xor(x, 1);
        x += __shfl_xor(x, 2);
        x += __shfl_xor(x, 4);
        x += __shfl_xor(x, 8);
        s[v] = x;
    }
    if ((lane & 15) == 0) {
        float4 o0, o1;
        const float bb = bc3[0];
        o0.x = 1.f / (1.f + expf(-(s[0] + bb)));
        o0.y = 1.f / (1.f + expf(-(s[1] + bb)));
        o0.z = 1.f / (1.f + expf(-(s[2] + bb)));
        o0.w = 1.f / (1.f + expf(-(s[3] + bb)));
        o1.x = 1.f / (1.f + expf(-(s[4] + bb)));
        o1.y = 1.f / (1.f + expf(-(s[5] + bb)));
        o1.z = 1.f / (1.f + expf(-(s[6] + bb)));
        o1.w = 1.f / (1.f + expf(-(s[7] + bb)));
        float* dst = out + ((size_t)(b * 512 + i)) * 512
                         + jt * 64 + wave * 16 + 8 * hh;
        *(float4*)(dst)     = o0;
        *(float4*)(dst + 4) = o1;
    }
}

// ---------------------------------------------------------------------------
extern "C" void kernel_launch(void* const* d_in, const int* in_sizes, int n_in,
                              void* d_out, int out_size, void* d_ws, size_t ws_size,
                              hipStream_t stream)
{
    const float* lnode = (const float*)d_in[0];
    const float* ledge = (const float*)d_in[1];
    const float* rnode = (const float*)d_in[2];
    const float* redge = (const float*)d_in[3];
    const float* WN  = (const float*)d_in[4];
    const float* bN  = (const float*)d_in[5];
    const float* WE  = (const float*)d_in[6];
    const float* bE  = (const float*)d_in[7];
    const float* Wp  = (const float*)d_in[8];
    const float* bp  = (const float*)d_in[9];
    const float* Wc1 = (const float*)d_in[10];
    const float* bc1 = (const float*)d_in[11];
    const float* Wc2 = (const float*)d_in[12];
    const float* bc2 = (const float*)d_in[13];
    const float* Wc3 = (const float*)d_in[14];
    const float* bc3 = (const float*)d_in[15];

    char* base = (char*)d_ws;
    size_t off = 0;
    auto alloc = [&](size_t bytes) -> char* {
        char* p = base + off;
        off = (off + bytes + 255) & ~(size_t)255;
        return p;
    };
    float*  lig = (float*)alloc((size_t)4 * 512 * 128 * 4);
    float*  rec = (float*)alloc((size_t)4 * 512 * 128 * 4);
    float*  plp = (float*)alloc((size_t)4 * 512 * 32 * 4);
    float*  prr = (float*)alloc((size_t)4 * 512 * 32 * 4);
    __bf16* wb1 = (__bf16*)alloc((size_t)WB_ELEMS * 2);
    __bf16* wb2 = (__bf16*)alloc((size_t)WB_ELEMS * 2);
    __bf16* x1  = (__bf16*)alloc((size_t)4 * 512 * 512 * 32 * 2);
    float*  out = (float*)d_out;

    gnn_kernel<<<4096, 128, 0, stream>>>(lnode, ledge, rnode, redge,
                                         WN, bN, WE, bE, lig, rec);
    pair_kernel<<<1024, 128, 0, stream>>>(lig, rec, Wp, bp, plp, prr);
    prep_w<<<(WB_ELEMS + 255) / 256, 256, 0, stream>>>(Wc1, wb1);
    prep_w<<<(WB_ELEMS + 255) / 256, 256, 0, stream>>>(Wc2, wb2);
    conv1_kernel<<<dim3(8, 512, 4), 128, 0, stream>>>(plp, prr, x1, wb1, bc1);
    conv2_kernel<<<dim3(8, 512, 4), 128, 0, stream>>>(x1, out, wb2, bc2, Wc3, bc3);
}
